// GCN_21655225107208
// MI455X (gfx1250) — matrix operand order repaired and checked by reading, not compiled, not run on hardware
//
#include <hip/hip_runtime.h>

typedef __attribute__((ext_vector_type(16))) _Float16 v16h;
typedef __attribute__((ext_vector_type(8)))  _Float16 v8h;
typedef __attribute__((ext_vector_type(4)))  _Float16 v4h;
typedef __attribute__((ext_vector_type(2)))  _Float16 v2h;
typedef __attribute__((ext_vector_type(8)))  float    v8f;
typedef __attribute__((ext_vector_type(4)))  float    v4f;

static inline int cdiv(long a, long b) { return (int)((a + b - 1) / b); }
static inline size_t alignup(size_t x) { return (x + 255) & ~(size_t)255; }

// ---------------- degree / norm ----------------
__global__ __launch_bounds__(256) void k_deg_init(float* __restrict__ deg, int N) {
  int i = blockIdx.x * 256 + threadIdx.x;
  if (i < N) deg[i] = 1.0f;              // self-loop contributes 1
}

__global__ __launch_bounds__(256) void k_deg_count(const int* __restrict__ dst,
                                                   float* __restrict__ deg, int E) {
  int i = blockIdx.x * 256 + threadIdx.x;
  if (i < E) atomicAdd(&deg[dst[i]], 1.0f);
}

__global__ __launch_bounds__(256) void k_deg_rsqrt(float* __restrict__ deg, int N) {
  int i = blockIdx.x * 256 + threadIdx.x;
  if (i < N) deg[i] = rsqrtf(deg[i]);    // dis = deg^-1/2
}

// ---- contiguous 8-element loads (f32 -> f16 convert, or straight f16) ----
__device__ __forceinline__ v8h load8(const float* __restrict__ p) {
  v4f a = *(const v4f*)p;
  v4f b = *(const v4f*)(p + 4);
  v8h r;
#pragma unroll
  for (int i = 0; i < 4; ++i) { r[i] = (_Float16)a[i]; r[i + 4] = (_Float16)b[i]; }
  return r;
}
__device__ __forceinline__ v8h load8(const _Float16* __restrict__ p) {
  return *(const v8h*)p;
}

// ---------------- WMMA GEMM: C(f16) = A(MxK) * B(KxNcols), f32 accumulate ----------------
// Each wave owns one 16-col N-tile, preloads ALL B fragments (K/32) into registers once,
// then streams M-tiles with incremented base pointers (constant store offsets).
template <typename TA, int K, int NT>   // NT = Ncols/16
__global__ __launch_bounds__(256) void k_gemm_wmma(
    const TA* __restrict__ A, const float* __restrict__ B,
    _Float16* __restrict__ C, int M)
{
  constexpr int Ncols = NT * 16;
  constexpr int KT = K / 32;
  const int wave = threadIdx.x >> 5;
  const int lane = threadIdx.x & 31;
  const int gw   = blockIdx.x * 8 + wave;          // global wave id
  const int nt      = gw % NT;
  const int mstart  = gw / NT;
  const int mstride = (gridDim.x * 8) / NT;

  const int hi  = lane >> 4;                        // 0: lanes 0-15, 1: lanes 16-31
  const int l15 = lane & 15;
  const int ncol = nt * 16 + l15;

  // Preload B fragments for whole K (B 32x16 f16 layout: column N=l15;
  // elements 0..7 hold K=kt*32+hi*8+0..7, elements 8..15 hold +16).
  v16h bf[KT];
#pragma unroll
  for (int kt = 0; kt < KT; ++kt)
#pragma unroll
    for (int h = 0; h < 16; ++h)
      bf[kt][h] = (_Float16)B[(size_t)(kt * 32 + (hi << 3) + (h & 7) + ((h >> 3) << 4)) * Ncols + ncol];

  const int mtiles = (M + 15) >> 4;                 // total M tiles (last may be partial)
  const int full   = M >> 4;                        // tiles with all 16 rows in range

  // Incrementally-advanced pointers: no per-iteration index recomputation.
  const TA* Ap = A + (size_t)(mstart * 16 + l15) * K;
  _Float16* Cp = C + (size_t)(mstart * 16 + hi * 8) * Ncols + ncol;
  const size_t Astep = (size_t)mstride * 16 * K;
  const size_t Cstep = (size_t)mstride * 16 * Ncols;

  int mt = mstart;
  for (; mt < full; mt += mstride, Ap += Astep, Cp += Cstep) {
    v8f acc = {};
#pragma unroll
    for (int kt = 0; kt < KT; ++kt) {
      // A 16x32 f16 layout: halves 0..7 -> K=kt*32+hi*8+0..7; halves 8..15 -> +16
      v8h lo  = load8(Ap + kt * 32 + hi * 8);
      v8h hiv = load8(Ap + kt * 32 + 16 + hi * 8);
      v16h a;
#pragma unroll
      for (int i = 0; i < 8; ++i) { a[i] = lo[i]; a[i + 8] = hiv[i]; }
      acc = __builtin_amdgcn_wmma_f32_16x16x32_f16(
          false, a, false, bf[kt], (short)0, acc, false, false);
    }
    // f32 C/D layout: VGPR r -> row mt*16 + r + hi*8; constant immediate offsets
#pragma unroll
    for (int r = 0; r < 8; ++r)
      Cp[(size_t)r * Ncols] = (_Float16)acc[r];
  }

  // Tail: at most one partial M-tile per wave (only when M % 16 != 0).
  if (mt < mtiles) {
    int mrow = mt * 16 + l15;
    if (mrow >= M) mrow = M - 1;                    // clamp: EXEC stays all-1 for WMMA
    const TA* Abase = A + (size_t)mrow * K;
    v8f acc = {};
#pragma unroll
    for (int kt = 0; kt < KT; ++kt) {
      v8h lo  = load8(Abase + kt * 32 + hi * 8);
      v8h hiv = load8(Abase + kt * 32 + 16 + hi * 8);
      v16h a;
#pragma unroll
      for (int i = 0; i < 8; ++i) { a[i] = lo[i]; a[i + 8] = hiv[i]; }
      acc = __builtin_amdgcn_wmma_f32_16x16x32_f16(
          false, a, false, bf[kt], (short)0, acc, false, false);
    }
#pragma unroll
    for (int r = 0; r < 8; ++r) {
      const int m = mt * 16 + r + hi * 8;
      if (m < M) C[(size_t)m * Ncols + ncol] = (_Float16)acc[r];
    }
  }
}

// ---------------- self-loop init: accum[v][f] = dis[v]^2 * feat[v][f] ----------------
template <int F>
__global__ __launch_bounds__(256) void k_selfinit(
    const float* __restrict__ dis, const _Float16* __restrict__ feat,
    float* __restrict__ accum, int N)
{
  const int t = blockIdx.x * 256 + threadIdx.x;
  if (t >= N * F) return;
  const int v = t / F;
  const float dv = dis[v];
  accum[t] = dv * dv * (float)feat[t];
}

// ---------------- edge scatter: one wave per edge, 32 lanes x (F/32) feats ----------------
template <int F>
__global__ __launch_bounds__(256) void k_scatter(
    const int* __restrict__ src, const int* __restrict__ dst,
    const float* __restrict__ dis, const _Float16* __restrict__ feat,
    float* __restrict__ accum, int E)
{
  const int t = blockIdx.x * 256 + threadIdx.x;
  const int e = t >> 5;
  const int lane = t & 31;
  if (e >= E) return;
  const int s = src[e];
  const int d = dst[e];
  const float nrm = dis[s] * dis[d];
  constexpr int PER = F / 32;
  const _Float16* fr = feat + (size_t)s * F + lane * PER;
  float* ar = accum + (size_t)d * F + lane * PER;
  if constexpr (PER == 4) {
    v4h v = *(const v4h*)fr;                        // single b64 load
#pragma unroll
    for (int i = 0; i < 4; ++i) atomicAdd(&ar[i], nrm * (float)v[i]);
  } else {
    v2h v = *(const v2h*)fr;                        // single b32 load
#pragma unroll
    for (int i = 0; i < 2; ++i) atomicAdd(&ar[i], nrm * (float)v[i]);
  }
}

// ---------------- bias + PReLU ----------------
template <int F>
__global__ __launch_bounds__(256) void k_act_to_f16(
    const float* __restrict__ accum, const float* __restrict__ bias,
    const float* __restrict__ slope, _Float16* __restrict__ out, int total)
{
  const int t = blockIdx.x * 256 + threadIdx.x;
  if (t >= total) return;
  const float v = accum[t] + bias[t & (F - 1)];
  const float sa = *slope;
  out[t] = (_Float16)(v >= 0.f ? v : sa * v);
}

template <int F>
__global__ __launch_bounds__(256) void k_act_inplace(
    float* __restrict__ accum, const float* __restrict__ bias,
    const float* __restrict__ slope, int total)
{
  const int t = blockIdx.x * 256 + threadIdx.x;
  if (t >= total) return;
  const float v = accum[t] + bias[t & (F - 1)];
  const float sa = *slope;
  accum[t] = (v >= 0.f ? v : sa * v);
}

extern "C" void kernel_launch(void* const* d_in, const int* in_sizes, int n_in,
                              void* d_out, int out_size, void* d_ws, size_t ws_size,
                              hipStream_t stream)
{
  (void)n_in; (void)out_size; (void)ws_size;
  const float* x  = (const float*)d_in[0];
  const int*   ei = (const int*)d_in[1];
  const float* W1 = (const float*)d_in[2];
  const float* b1 = (const float*)d_in[3];
  const float* W2 = (const float*)d_in[4];
  const float* b2 = (const float*)d_in[5];
  const float* a  = (const float*)d_in[6];

  const int IN = 128, HID = 128, OUT = 64;
  const int N = in_sizes[0] / IN;
  const int E = in_sizes[1] / 2;
  const int* src = ei;
  const int* dst = ei + E;

  // workspace layout
  char* ws = (char*)d_ws;
  float*    dis  = (float*)ws;        ws += alignup((size_t)N * 4);
  _Float16* f16a = (_Float16*)ws;     ws += alignup((size_t)N * HID * 2); // hw16; reused for layer-2 gemm out
  float*    hacc = (float*)ws;        ws += alignup((size_t)N * HID * 4); // f32 accumulator, layer 1
  _Float16* h16  = (_Float16*)ws;     ws += alignup((size_t)N * HID * 2); // activated layer-1 features
  float*    outp = (float*)d_out;

  // --- degrees / symmetric norm ---
  k_deg_init <<<cdiv(N, 256), 256, 0, stream>>>(dis, N);
  k_deg_count<<<cdiv(E, 256), 256, 0, stream>>>(dst, dis, E);
  k_deg_rsqrt<<<cdiv(N, 256), 256, 0, stream>>>(dis, N);

  // --- layer 1: hw16 = f16(x @ W1), WMMA ---
  k_gemm_wmma<float, 128, 8><<<256, 256, 0, stream>>>(x, W1, f16a, N);

  // --- aggregate: h = dis^2*hw + sum_edges norm*hw[src] ---
  k_selfinit<128><<<cdiv((long)N * 128, 256), 256, 0, stream>>>(dis, f16a, hacc, N);
  k_scatter<128> <<<cdiv((long)E * 32, 256), 256, 0, stream>>>(src, dst, dis, f16a, hacc, E);

  // --- h16 = f16(prelu(h + b1)) ---
  k_act_to_f16<128><<<cdiv((long)N * 128, 256), 256, 0, stream>>>(hacc, b1, a, h16, N * HID);

  // --- layer 2: hw2_16 = f16(h16 @ W2), WMMA (reuse f16a buffer) ---
  k_gemm_wmma<_Float16, 128, 4><<<256, 256, 0, stream>>>(h16, W2, f16a, N);

  // --- aggregate into d_out ---
  k_selfinit<64><<<cdiv((long)N * 64, 256), 256, 0, stream>>>(dis, f16a, outp, N);
  k_scatter<64> <<<cdiv((long)E * 32, 256), 256, 0, stream>>>(src, dst, dis, f16a, outp, E);

  // --- out = prelu(out + b2) in place ---
  k_act_inplace<64><<<cdiv((long)N * 64, 256), 256, 0, stream>>>(outp, b2, a, N * OUT);
}
